// ChannelCrossAttention_5961414607180
// MI455X (gfx1250) — compile-verified, matrix operand order checked by reading, not run on hardware
//
#include <hip/hip_runtime.h>

// ---------------------------------------------------------------------------
// ChannelCrossAttention on MI455X (gfx1250, wave32, WMMA)
//
// Pipeline (all bf16 matrix math, f32 accumulate):
//   0) transpose+convert x (B,C,N) f32 -> xT (B,N,C) bf16   [LDS-tiled]
//   1) convert wq/wk/wv f32 -> bf16
//   2) q = Wq*x + bq  -> (C,N) bf16      \
//      k = Wk*x + bk  -> (C,N) bf16       > generic WMMA GEMM kernel
//      v = Wv*x + bv  -> vT (N,C) bf16   /
//   3) energy = q k^T / sqrt(N) -> f32 (C,C)   [same GEMM kernel, K=16384]
//   4) softmax rows -> att bf16 (C,C)
//   5) out = att*v + x -> f32 (C,N)            [same GEMM kernel + residual]
//
// GEMM geometry: block = 8 waves (wave32) = 4(M) x 2(N); block tile 128x128;
// wave tile 32x64 = 2x4 fragments -> 8 v_wmma per K-step from 6 fragment
// loads. K-loop is explicitly double-buffered: next step's 12 b128 loads are
// issued before the current step's 8 WMMAs so s_wait_loadcnt covers a full
// WMMA burst. Epilogue variants are compile-time template parameters.
// ---------------------------------------------------------------------------

typedef __attribute__((ext_vector_type(16))) __bf16 v16bf;
typedef __attribute__((ext_vector_type(8)))  __bf16 v8bf;
typedef __attribute__((ext_vector_type(8)))  float  v8f;

#define BDIM 4
#define CDIM 256
#define NDIM 16384   // 128*128

// ---------------------------------------------------------------------------
// Fragment loader: 16-bit operand, row-major source, per-lane row index,
// contiguous K. Matches CDNA5 16-bit A-matrix 16x32 layout (ISA 7.12.2):
//   lanes 0-15 : row = lane,    VGPR0-3 K=0..7,  VGPR4-7 K=16..23
//   lanes 16-31: row = lane-16, VGPR0-3 K=8..15, VGPR4-7 K=24..31
// B operand (32x16) uses the symmetric layout with "row" = output column.
// ---------------------------------------------------------------------------
__device__ __forceinline__ v16bf load_frag(const __bf16* __restrict__ base,
                                           int row0, int ld, int k0) {
  int lane = threadIdx.x & 31;
  int r = lane & 15;
  int h = lane >> 4;
  const __bf16* p = base + (size_t)(row0 + r) * ld + k0 + h * 8;
  v8bf lo = *(const v8bf*)(p);
  v8bf hi = *(const v8bf*)(p + 16);
  v16bf f;
#pragma unroll
  for (int i = 0; i < 8; ++i) { f[i] = lo[i]; f[i + 8] = hi[i]; }
  return f;
}

#define WMMA_BF16(A_, B_, C_) \
  __builtin_amdgcn_wmma_f32_16x16x32_bf16(false, (A_), false, (B_), (short)0, (C_), false, false)

// ---------------------------------------------------------------------------
// Generic WMMA GEMM: D[m,n] = alpha * sum_k A[m,k]*Bop[n,k] (+bias[m]) (+resid)
//   A  : row-major (lda), per-lane contiguous in K
//   Bop: "column-major view" — row index is the output column n, contiguous K
// ---------------------------------------------------------------------------
template <bool HAS_BIAS, bool HAS_RESID, bool OUT_F32, bool TRANS>
__global__ __launch_bounds__(256) void wmma_gemm_bf16(
    const __bf16* __restrict__ A, int lda, long long strideA,
    const __bf16* __restrict__ Bop, int ldb, long long strideB,
    const float* __restrict__ bias,
    float alpha,
    const float* __restrict__ resid, long long strideR,
    float* __restrict__ outF,
    __bf16* __restrict__ outBf,
    int ldo, long long strideO,
    int Kdim) {
  int b = blockIdx.z;
  A   += strideA * b;
  Bop += strideB * b;
  if (HAS_RESID) resid += strideR * b;
  if (OUT_F32)   outF  += strideO * b;
  else           outBf += strideO * b;

  int wave = threadIdx.x >> 5;   // 0..7
  int lane = threadIdx.x & 31;
  int wm = wave >> 1;            // 0..3 -> M sub-tile (32 rows)
  int wn = wave & 1;             // 0..1 -> N sub-tile (64 cols)
  int m0 = blockIdx.x * 128 + wm * 32;
  int n0 = blockIdx.y * 128 + wn * 64;

  v8f z = {0.f, 0.f, 0.f, 0.f, 0.f, 0.f, 0.f, 0.f};
  v8f acc[2][4];
#pragma unroll
  for (int i = 0; i < 2; ++i)
#pragma unroll
    for (int j = 0; j < 4; ++j) acc[i][j] = z;

  // ---- software-pipelined K loop (1-deep double buffer) ----
  v16bf a0 = load_frag(A, m0,      lda, 0);
  v16bf a1 = load_frag(A, m0 + 16, lda, 0);
  v16bf b0 = load_frag(Bop, n0,      ldb, 0);
  v16bf b1 = load_frag(Bop, n0 + 16, ldb, 0);
  v16bf b2 = load_frag(Bop, n0 + 32, ldb, 0);
  v16bf b3 = load_frag(Bop, n0 + 48, ldb, 0);

  for (int k0 = 0; k0 < Kdim; k0 += 32) {
    int kn = (k0 + 32 < Kdim) ? (k0 + 32) : 0;   // wrapped prefetch (in-bounds)
    v16bf na0 = load_frag(A, m0,      lda, kn);
    v16bf na1 = load_frag(A, m0 + 16, lda, kn);
    v16bf nb0 = load_frag(Bop, n0,      ldb, kn);
    v16bf nb1 = load_frag(Bop, n0 + 16, ldb, kn);
    v16bf nb2 = load_frag(Bop, n0 + 32, ldb, kn);
    v16bf nb3 = load_frag(Bop, n0 + 48, ldb, kn);

    acc[0][0] = WMMA_BF16(a0, b0, acc[0][0]);
    acc[1][0] = WMMA_BF16(a1, b0, acc[1][0]);
    acc[0][1] = WMMA_BF16(a0, b1, acc[0][1]);
    acc[1][1] = WMMA_BF16(a1, b1, acc[1][1]);
    acc[0][2] = WMMA_BF16(a0, b2, acc[0][2]);
    acc[1][2] = WMMA_BF16(a1, b2, acc[1][2]);
    acc[0][3] = WMMA_BF16(a0, b3, acc[0][3]);
    acc[1][3] = WMMA_BF16(a1, b3, acc[1][3]);

    a0 = na0; a1 = na1;
    b0 = nb0; b1 = nb1; b2 = nb2; b3 = nb3;
  }

  // Epilogue. C/D layout (ISA 7.12.2): VGPR r, lane L -> M = r + 8*(L>=16),
  // N = L&15 within the 16x16 fragment.
  int r15 = lane & 15;
  int h8  = (lane >> 4) * 8;
#pragma unroll
  for (int i = 0; i < 2; ++i) {
#pragma unroll
    for (int j = 0; j < 4; ++j) {
      int mb = m0 + i * 16 + h8;
      int n  = n0 + j * 16 + r15;
#pragma unroll
      for (int r = 0; r < 8; ++r) {
        int m = mb + r;
        float val = acc[i][j][r] * alpha;
        if (HAS_BIAS)  val += bias[m];
        if (HAS_RESID) val += resid[(size_t)m * ldo + n];
        if (OUT_F32) {
          outF[(size_t)m * ldo + n] = val;
        } else if (TRANS) {
          outBf[(size_t)n * ldo + m] = (__bf16)val;   // ldo = row length of T
        } else {
          outBf[(size_t)m * ldo + n] = (__bf16)val;
        }
      }
    }
  }
}

// ---------------------------------------------------------------------------
// x (B,C,N) f32 -> xT (B,N,C) bf16, LDS-tiled 32x32 transpose
// ---------------------------------------------------------------------------
__global__ __launch_bounds__(256) void transpose_to_bf16(
    const float* __restrict__ x, __bf16* __restrict__ xT) {
  __shared__ __bf16 tile[32][33];
  int b  = blockIdx.z;
  int n0 = blockIdx.x * 32;
  int c0 = blockIdx.y * 32;
  int tx = threadIdx.x;   // 0..31
  int ty = threadIdx.y;   // 0..7
  const float* xb = x + (size_t)b * CDIM * NDIM;
#pragma unroll
  for (int i = 0; i < 32; i += 8) {
    int c = c0 + ty + i;
    tile[ty + i][tx] = (__bf16)xb[(size_t)c * NDIM + n0 + tx];
  }
  __syncthreads();
  __bf16* xTb = xT + (size_t)b * NDIM * CDIM;
#pragma unroll
  for (int i = 0; i < 32; i += 8) {
    int n = n0 + ty + i;
    xTb[(size_t)n * CDIM + c0 + tx] = tile[tx][ty + i];
  }
}

// ---------------------------------------------------------------------------
// wq/wk/wv f32 -> bf16 (contiguous: [wq_bf | wk_bf | wv_bf])
// ---------------------------------------------------------------------------
__global__ __launch_bounds__(256) void convert_weights(
    const float* __restrict__ wq, const float* __restrict__ wk,
    const float* __restrict__ wv, __bf16* __restrict__ out) {
  int i = blockIdx.x * 256 + threadIdx.x;     // 0 .. 3*65536-1
  int which = i >> 16;
  int off   = i & 0xFFFF;
  const float* s = (which == 0) ? wq : (which == 1) ? wk : wv;
  out[i] = (__bf16)s[off];
}

// ---------------------------------------------------------------------------
// Row softmax: energy f32 (B*C rows of C) -> att bf16
// ---------------------------------------------------------------------------
__global__ __launch_bounds__(256) void softmax_rows(
    const float* __restrict__ energy, __bf16* __restrict__ att) {
  __shared__ float red[256];
  int row = blockIdx.x;                        // b*C + c
  int t = threadIdx.x;
  float v = energy[(size_t)row * CDIM + t];
  red[t] = v; __syncthreads();
#pragma unroll
  for (int s = 128; s > 0; s >>= 1) {
    if (t < s) red[t] = fmaxf(red[t], red[t + s]);
    __syncthreads();
  }
  float m = red[0]; __syncthreads();
  float ex = __expf(v - m);
  red[t] = ex; __syncthreads();
#pragma unroll
  for (int s = 128; s > 0; s >>= 1) {
    if (t < s) red[t] += red[t + s];
    __syncthreads();
  }
  float sum = red[0];
  att[(size_t)row * CDIM + t] = (__bf16)(ex / sum);
}

// ---------------------------------------------------------------------------
extern "C" void kernel_launch(void* const* d_in, const int* in_sizes, int n_in,
                              void* d_out, int out_size, void* d_ws, size_t ws_size,
                              hipStream_t stream) {
  (void)in_sizes; (void)n_in; (void)out_size; (void)ws_size;
  const float* x  = (const float*)d_in[0];
  const float* wq = (const float*)d_in[1];
  const float* bq = (const float*)d_in[2];
  const float* wk = (const float*)d_in[3];
  const float* bk = (const float*)d_in[4];
  const float* wv = (const float*)d_in[5];
  const float* bv = (const float*)d_in[6];
  float* out = (float*)d_out;

  const size_t CN = (size_t)CDIM * NDIM;
  char* p = (char*)d_ws;
  __bf16* xT  = (__bf16*)p; p += (size_t)BDIM * CN * sizeof(__bf16);  // (B,N,C)
  __bf16* qb  = (__bf16*)p; p += (size_t)BDIM * CN * sizeof(__bf16);  // (B,C,N)
  __bf16* kb  = (__bf16*)p; p += (size_t)BDIM * CN * sizeof(__bf16);  // (B,C,N)
  __bf16* vT  = (__bf16*)p; p += (size_t)BDIM * CN * sizeof(__bf16);  // (B,N,C)
  __bf16* wbf = (__bf16*)p; p += (size_t)3 * CDIM * CDIM * sizeof(__bf16);
  float*  eng = (float*)p;  p += (size_t)BDIM * CDIM * CDIM * sizeof(float);
  __bf16* att = (__bf16*)p;
  __bf16* wqb = wbf;
  __bf16* wkb = wbf + CDIM * CDIM;
  __bf16* wvb = wbf + 2 * CDIM * CDIM;

  // 0) x -> xT bf16
  transpose_to_bf16<<<dim3(NDIM / 32, CDIM / 32, BDIM), dim3(32, 8), 0, stream>>>(x, xT);
  // 1) weights -> bf16
  convert_weights<<<dim3(3 * CDIM * CDIM / 256), dim3(256), 0, stream>>>(wq, wk, wv, wbf);

  dim3 gBig(CDIM / 128, NDIM / 128, BDIM);   // (2, 128, 4)
  dim3 blk(256);

  // 2) q, k: (C,N) bf16 ; v: transposed -> vT (N,C) bf16
  wmma_gemm_bf16<true, false, false, false><<<gBig, blk, 0, stream>>>(
      wqb, CDIM, 0, xT, CDIM, (long long)CN, bq, 1.0f, nullptr, 0,
      nullptr, qb, NDIM, (long long)CN, CDIM);
  wmma_gemm_bf16<true, false, false, false><<<gBig, blk, 0, stream>>>(
      wkb, CDIM, 0, xT, CDIM, (long long)CN, bk, 1.0f, nullptr, 0,
      nullptr, kb, NDIM, (long long)CN, CDIM);
  wmma_gemm_bf16<true, false, false, true><<<gBig, blk, 0, stream>>>(
      wvb, CDIM, 0, xT, CDIM, (long long)CN, bv, 1.0f, nullptr, 0,
      nullptr, vT, CDIM, (long long)CN, CDIM);

  // 3) energy = q k^T / sqrt(N)  (f32, per batch 256x256, K = 16384)
  dim3 gEng(CDIM / 128, CDIM / 128, BDIM);   // (2, 2, 4)
  wmma_gemm_bf16<false, false, true, false><<<gEng, blk, 0, stream>>>(
      qb, NDIM, (long long)CN, kb, NDIM, (long long)CN, nullptr, 1.0f / 128.0f,
      nullptr, 0, eng, nullptr, CDIM, (long long)(CDIM * CDIM), NDIM);

  // 4) softmax rows -> att bf16
  softmax_rows<<<dim3(BDIM * CDIM), dim3(256), 0, stream>>>(eng, att);

  // 5) out = att * v + x  (f32)
  wmma_gemm_bf16<false, true, true, false><<<gBig, blk, 0, stream>>>(
      att, CDIM, (long long)(CDIM * CDIM), vT, CDIM, (long long)CN, nullptr,
      1.0f, x, (long long)CN, out, nullptr, NDIM, (long long)CN, CDIM);
}
